// LinearAttention_79147657331088
// MI455X (gfx1250) — matrix operand order repaired and checked by reading, not compiled, not run on hardware
//
#include <hip/hip_runtime.h>
#include <hip/hip_bf16.h>
#include <math.h>

// ---------------------------------------------------------------------------
// Linear attention (elu+1 feature map), MI455X / gfx1250, wave32 + WMMA f16.
//
//   KV[n,h]  = sum_s K[n,s,h,:]^T * V[n,s,h,:]          (64x64 per head)
//   Ksum[n,h]= sum_s K[n,s,h,:]                          (64 per head)
//   Z        = 1 / (Q . Ksum + eps)
//   out      = (Q @ KV) * Z          (the /S ... *S of the reference cancels)
//
// Both matmuls run on v_wmma_f32_16x16x32_f16 (f16 in, f32 accumulate).
// LDS operands are stored in fragment-natural (transposed) order so each
// WMMA fragment is exactly two aligned ds_load_b128s; rows are padded so the
// 16 lanes of each half-wave hit 16 distinct LDS banks.
// ---------------------------------------------------------------------------

typedef __attribute__((ext_vector_type(16))) _Float16 v16h;
typedef __attribute__((ext_vector_type(8)))  _Float16 v8h;
typedef __attribute__((ext_vector_type(4)))  _Float16 v4h;
typedef __attribute__((ext_vector_type(8)))  float    v8f;

constexpr int N_  = 8;
constexpr int L_  = 4096;
constexpr int S_  = 4096;
constexpr int H_  = 8;
constexpr int D_  = 64;   // feature dim
constexpr int DV_ = 64;   // value dim
constexpr int NH_ = N_ * H_;

constexpr int SPLIT_  = 16;            // S-splits per (n,h) in phase 1
constexpr int SCHUNK_ = S_ / SPLIT_;   // 256 rows of S per block
constexpr int ITERS_  = SCHUNK_ / 32;  // 32-row LDS chunks -> 8 iterations

constexpr int P1PAD_ = 40;  // 32 s-values padded to 40 halves (80B rows: 16B
                            // aligned, 20-dword stride, gcd(20,64)=4 -> 16
                            // lanes hit 16 distinct banks)
constexpr int P2PAD_ = 72;  // 64 halves padded to 72 (144B rows: 16B aligned,
                            // 36-dword stride, conflict-free for 16 lanes)

__device__ __forceinline__ float elu1(float x) {
    // elu(x) + 1  ==  x+1 (x>0)  |  exp(x) (x<=0)
    return x > 0.0f ? x + 1.0f : __expf(x);
}

__device__ __forceinline__ v16h join8(v8h lo, v8h hi) {
    return __builtin_shufflevector(lo, hi, 0, 1, 2, 3, 4, 5, 6, 7,
                                           8, 9, 10, 11, 12, 13, 14, 15);
}

__global__ void __launch_bounds__(256) zero_ws_kernel(float* p, int n) {
    int i = blockIdx.x * 256 + threadIdx.x;
    if (i < n) p[i] = 0.0f;
}

// ---------------------------------------------------------------------------
// Phase 1: KV = K^T V  and  Ksum = sum K, merged with f32 global atomics.
// grid = NH_*SPLIT_, block = 256 (8 waves). Each wave owns 2 of the 16
// (d-tile, v-tile) 16x16 output tiles of the 64x64 KV state.
// ---------------------------------------------------------------------------
__global__ void __launch_bounds__(256)
linattn_kv_kernel(const float* __restrict__ kg, const float* __restrict__ vg,
                  float* __restrict__ kvws, float* __restrict__ ksumws) {
    // Transposed staging: Kt[d][s], Vt[n][s] -> fragment reads are contiguous
    __shared__ _Float16 Kt[D_][P1PAD_];
    __shared__ _Float16 Vt[DV_][P1PAD_];

    const int tid  = threadIdx.x;
    const int lane = tid & 31;
    const int wave = tid >> 5;
    const int half = lane >> 4;       // 0: lanes 0-15, 1: lanes 16-31
    const int ln16 = lane & 15;

    const int nh    = blockIdx.x / SPLIT_;
    const int chunk = blockIdx.x % SPLIT_;
    const int n     = nh / H_;
    const int h     = nh % H_;
    const int sbase = chunk * SCHUNK_;

    // wave -> (d-tile, two v-tiles); 8 waves cover the 4x4 tile grid
    const int dt  = wave >> 1;
    const int vt0 = (wave & 1) * 2;
    const int vt1 = vt0 + 1;
    const int md  = dt * 16 + ln16;   // A-fragment row (feature dim)
    const int nc0 = vt0 * 16 + ln16;  // B-fragment cols (value dims)
    const int nc1 = vt1 * 16 + ln16;

    v8f c0 = {0.f, 0.f, 0.f, 0.f, 0.f, 0.f, 0.f, 0.f};
    v8f c1 = {0.f, 0.f, 0.f, 0.f, 0.f, 0.f, 0.f, 0.f};
    float kacc = 0.0f;

    for (int it = 0; it < ITERS_; ++it) {
        __syncthreads();   // protect LDS from previous iteration's readers
        // --- stage 32 rows of K (featurized) and V, f32x4 global loads,
        //     transposed f16 scatter into LDS ---
        #pragma unroll
        for (int rep = 0; rep < 2; ++rep) {
            int idx = (rep * 256 + tid) * 4;    // 0 .. 2044, 4-aligned
            int row = idx >> 6;                 // s within chunk: 0..31
            int col = idx & 63;                 // d0 (multiple of 4)
            size_t g = ((size_t)(n * S_ + sbase + it * 32 + row) * H_ + h)
                         * (size_t)D_ + col;
            float4 kf = *(const float4*)&kg[g];
            float4 vf = *(const float4*)&vg[g];
            Kt[col + 0][row] = (_Float16)elu1(kf.x);
            Kt[col + 1][row] = (_Float16)elu1(kf.y);
            Kt[col + 2][row] = (_Float16)elu1(kf.z);
            Kt[col + 3][row] = (_Float16)elu1(kf.w);
            Vt[col + 0][row] = (_Float16)vf.x;
            Vt[col + 1][row] = (_Float16)vf.y;
            Vt[col + 2][row] = (_Float16)vf.z;
            Vt[col + 3][row] = (_Float16)vf.w;
            if (it + 1 < ITERS_) {
                __builtin_prefetch(&kg[g + (size_t)32 * H_ * D_], 0, 1);
                __builtin_prefetch(&vg[g + (size_t)32 * H_ * D_], 0, 1);
            }
        }
        __syncthreads();

        // --- Ksum partial (one dim per thread; contiguous f16 row) ---
        if (tid < D_) {
            #pragma unroll
            for (int s = 0; s < 32; ++s) kacc += (float)Kt[tid][s];
        }

        // --- A fragment (K^T, 16x32): lanes 0-15 K=0-15; lanes 16-31
        //     K=16-31 -> two contiguous v8h chunks per lane ---
        v16h a = join8(*(const v8h*)&Kt[md][half * 16],
                       *(const v8h*)&Kt[md][half * 16 + 8]);
        // --- B fragments (V, 32x16): lane = N; lanes 0-15 K=0-15, lanes
        //     16-31 K=16-31 -> two contiguous v8h chunks per lane ---
        v16h b0 = join8(*(const v8h*)&Vt[nc0][half * 16],
                        *(const v8h*)&Vt[nc0][half * 16 + 8]);
        v16h b1 = join8(*(const v8h*)&Vt[nc1][half * 16],
                        *(const v8h*)&Vt[nc1][half * 16 + 8]);

        c0 = __builtin_amdgcn_wmma_f32_16x16x32_f16(false, a, false, b0,
                                                    (short)0, c0, false, false);
        c1 = __builtin_amdgcn_wmma_f32_16x16x32_f16(false, a, false, b1,
                                                    (short)0, c1, false, false);
    }

    // --- merge partials into the per-head state ---
    if (tid < D_) atomicAdd(&ksumws[nh * D_ + tid], kacc);

    // C layout: VGPR r -> M = r (lanes 0-15) / M = r+8 (lanes 16-31)
    float* kvbase = kvws + (size_t)nh * D_ * DV_;
    #pragma unroll
    for (int r = 0; r < 8; ++r) {
        int m = dt * 16 + r + (half ? 8 : 0);
        atomicAdd(&kvbase[m * DV_ + nc0], c0[r]);
        atomicAdd(&kvbase[m * DV_ + nc1], c1[r]);
    }
}

// ---------------------------------------------------------------------------
// Phase 2: out = (Q @ KV) * Z.  grid = NH_ * (L_/64), block = 256 (8 waves).
// Each block: one head, 64 query rows; each wave owns 2 of the 16 output tiles.
// ---------------------------------------------------------------------------
__global__ void __launch_bounds__(256)
linattn_out_kernel(const float* __restrict__ qg, const float* __restrict__ kvws,
                   const float* __restrict__ ksumws, float* __restrict__ outg) {
    __shared__ _Float16 Qh [64][P2PAD_];   // [l][d] row-major: A-natural
    __shared__ _Float16 KVt[DV_][P2PAD_];  // [v][d] transposed: B-natural
    __shared__ float    ksl[D_];
    __shared__ float    Zl[64];

    const int tid  = threadIdx.x;
    const int lane = tid & 31;
    const int wave = tid >> 5;
    const int half = lane >> 4;
    const int ln16 = lane & 15;

    const int nblk  = L_ / 64;
    const int nh    = blockIdx.x / nblk;
    const int lbase = (blockIdx.x % nblk) * 64;
    const int n     = nh / H_;
    const int h     = nh % H_;

    if (tid < D_) ksl[tid] = ksumws[nh * D_ + tid];

    // KV state -> transposed f16 LDS (f32x4 loads, b16 scatter)
    const float* kvbase = kvws + (size_t)nh * D_ * DV_;
    #pragma unroll
    for (int rep = 0; rep < 4; ++rep) {
        int idx = (rep * 256 + tid) * 4;    // linear over KV[d][v]
        int d   = idx >> 6;
        int v0  = idx & 63;
        float4 kv4 = *(const float4*)&kvbase[idx];
        KVt[v0 + 0][d] = (_Float16)kv4.x;
        KVt[v0 + 1][d] = (_Float16)kv4.y;
        KVt[v0 + 2][d] = (_Float16)kv4.z;
        KVt[v0 + 3][d] = (_Float16)kv4.w;
    }
    // Q tile (featurized) -> row-major f16 LDS (f32x4 loads, b64 stores)
    #pragma unroll
    for (int rep = 0; rep < 4; ++rep) {
        int idx = (rep * 256 + tid) * 4;
        int row = idx >> 6;
        int col = idx & 63;
        size_t g = ((size_t)(n * L_ + lbase + row) * H_ + h) * (size_t)D_ + col;
        float4 q4 = *(const float4*)&qg[g];
        v4h qh;
        qh[0] = (_Float16)elu1(q4.x);
        qh[1] = (_Float16)elu1(q4.y);
        qh[2] = (_Float16)elu1(q4.z);
        qh[3] = (_Float16)elu1(q4.w);
        *(v4h*)&Qh[row][col] = qh;
    }
    __syncthreads();

    // per-row normalizer Z = 1 / (Q . Ksum + eps)
    if (tid < 64) {
        float dot = 0.0f;
        #pragma unroll
        for (int d = 0; d < D_; ++d) dot += (float)Qh[tid][d] * ksl[d];
        Zl[tid] = 1.0f / (dot + 1e-6f);
    }
    __syncthreads();

    const int mt  = wave >> 1;
    const int vt0 = (wave & 1) * 2;
    const int vt1 = vt0 + 1;
    const int mq  = mt * 16 + ln16;
    const int nc0 = vt0 * 16 + ln16;
    const int nc1 = vt1 * 16 + ln16;

    v8f c0 = {0.f, 0.f, 0.f, 0.f, 0.f, 0.f, 0.f, 0.f};
    v8f c1 = {0.f, 0.f, 0.f, 0.f, 0.f, 0.f, 0.f, 0.f};

    #pragma unroll
    for (int ks = 0; ks < 2; ++ks) {           // contraction over D in 2x32
        const int kb = ks * 32;
        v16h a  = join8(*(const v8h*)&Qh[mq][kb + half * 16],
                        *(const v8h*)&Qh[mq][kb + half * 16 + 8]);
        v16h b0 = join8(*(const v8h*)&KVt[nc0][kb + half * 16],
                        *(const v8h*)&KVt[nc0][kb + half * 16 + 8]);
        v16h b1 = join8(*(const v8h*)&KVt[nc1][kb + half * 16],
                        *(const v8h*)&KVt[nc1][kb + half * 16 + 8]);
        c0 = __builtin_amdgcn_wmma_f32_16x16x32_f16(false, a, false, b0,
                                                    (short)0, c0, false, false);
        c1 = __builtin_amdgcn_wmma_f32_16x16x32_f16(false, a, false, b1,
                                                    (short)0, c1, false, false);
    }

    // scale rows by Z and store (16 contiguous Dv cols per half-wave row ->
    // coalesced b32 stores)
    #pragma unroll
    for (int r = 0; r < 8; ++r) {
        int m = mt * 16 + r + (half ? 8 : 0);
        float z = Zl[m];
        size_t g = ((size_t)(n * L_ + lbase + m) * H_ + h) * (size_t)DV_;
        outg[g + nc0] = c0[r] * z;
        outg[g + nc1] = c1[r] * z;
    }
}

// ---------------------------------------------------------------------------
extern "C" void kernel_launch(void* const* d_in, const int* in_sizes, int n_in,
                              void* d_out, int out_size, void* d_ws, size_t ws_size,
                              hipStream_t stream) {
    const float* q = (const float*)d_in[0];
    const float* k = (const float*)d_in[1];
    const float* v = (const float*)d_in[2];
    float* out = (float*)d_out;

    float* kvws   = (float*)d_ws;                   // NH * 64 * 64 f32 (1 MiB)
    float* ksumws = kvws + (size_t)NH_ * D_ * DV_;  // NH * 64 f32

    const int ws_elems = NH_ * D_ * DV_ + NH_ * D_;
    zero_ws_kernel<<<(ws_elems + 255) / 256, 256, 0, stream>>>(kvws, ws_elems);

    linattn_kv_kernel<<<NH_ * SPLIT_, 256, 0, stream>>>(k, v, kvws, ksumws);
    linattn_out_kernel<<<NH_ * (L_ / 64), 256, 0, stream>>>(q, kvws, ksumws, out);
}
